// EmbeddingOneHopAttention_38070590112248
// MI455X (gfx1250) — matrix operand order repaired and checked by reading, not compiled, not run on hardware
//
#include <hip/hip_runtime.h>
#include <hip/hip_bf16.h>

typedef __bf16 bf16;
typedef __attribute__((ext_vector_type(16))) __bf16 v16bf;
typedef __attribute__((ext_vector_type(8)))  float  v8f;

#define ES      128
#define KNB     50
#define TILE    16
#define THREADS 256
#define NPAIRS  4096   // B*N = 4*1024

__device__ __forceinline__ bf16 f2bf(float f) { return (bf16)f; }

// A-matrix fragment: 16x32 bf16 tile from LDS matrix M (row stride ES).
// ISA layout: lanes 0-15 hold M=lane, VGPR0-3 = K={k0..k0+7}, VGPR4-7 = K={k0+16..k0+23};
// lanes 16-31 same rows, K offset +8 within each half.
__device__ __forceinline__ v16bf load_a(const bf16* M, int k0, int lane) {
    v16bf a;
    const int m  = lane & 15;
    const int kh = (lane & 16) ? 8 : 0;
    const bf16* row = M + m * ES;
#pragma unroll
    for (int p = 0; p < 4; ++p) {
        int k = k0 + kh + 2 * p;
        a[2 * p]     = row[k];
        a[2 * p + 1] = row[k + 1];
        a[8 + 2 * p]     = row[k + 16];
        a[8 + 2 * p + 1] = row[k + 17];
    }
    return a;
}

// B-matrix fragment: 32x16 tile; B[k][n] = W[n][k] with W stored row-major [n*ES + k].
// ISA layout: lanes 0-15 -> N=lane, VGPR0-3 = K={k0..k0+7}, VGPR4-7 = K={k0+16..k0+23}; lanes 16-31 K offset +8.
__device__ __forceinline__ v16bf load_b(const bf16* W, int k0, int n0, int lane) {
    v16bf b;
    const int n  = n0 + (lane & 15);
    const int kh = (lane & 16) ? 8 : 0;
    const bf16* row = W + n * ES + k0 + kh;
#pragma unroll
    for (int p = 0; p < 8; ++p) {
        b[2 * p]     = row[2 * p + ((p & 4) ? 8 : 0)];
        b[2 * p + 1] = row[2 * p + 1 + ((p & 4) ? 8 : 0)];
    }
    return b;
}

__global__ void __launch_bounds__(THREADS)
koa_kernel(const float* __restrict__ ent_emb,
           const float* __restrict__ rel_emb,
           const float* __restrict__ W_bil,
           const float* __restrict__ W1,
           const float* __restrict__ b1,
           const float* __restrict__ W2,
           const float* __restrict__ b2,
           const int*   __restrict__ head_idx,
           const int*   __restrict__ tail_idx,
           const int*   __restrict__ conn_rel,
           float*       __restrict__ out)
{
    extern __shared__ __align__(16) char smem[];
    bf16*  sW   = (bf16*)smem;                 // 3 * 128*128 bf16 weights (W_bil, W1, W2)
    bf16*  sWR  = sW + 3 * ES * ES;            // weak_rel       [16][128] bf16
    bf16*  sHE  = sWR + TILE * ES;             // head entity    [16][128] bf16
    bf16*  sTE  = sHE + TILE * ES;             // tail entity    [16][128] bf16
    bf16*  sHA  = sTE + TILE * ES;             // head aggregate [16][128] bf16
    bf16*  sTA  = sHA + TILE * ES;             // tail aggregate [16][128] bf16
    float* sWS  = (float*)(sTA + TILE * ES);   // ws             [16][128] f32
    float* sATT = sWS + TILE * ES;             // per-wave attention [8][64]
    int*   sNB  = (int*)(sATT + 8 * 64);       // per-wave neighbor ids [8][64]
    int*   sIDX = sNB + 8 * 64;                // head idx [16], tail idx [16]

    const int t    = threadIdx.x;
    const int lane = t & 31;
    const int wave = t >> 5;
    const int tb   = blockIdx.x * TILE;        // first pair of this tile

    // ---------------- Phase 0: stage weights (bf16) + entity rows + weak_rel ----------------
    for (int i = t; i < ES * ES; i += THREADS) {
        sW[i]              = f2bf(W_bil[i]);
        sW[ES * ES + i]    = f2bf(W1[i]);
        sW[2 * ES * ES + i] = f2bf(W2[i]);
    }
    if (t < TILE) {
        sIDX[t]        = head_idx[tb + t];
        sIDX[TILE + t] = tail_idx[tb + t];
    }
    {
        const int p  = t >> 4;                  // pair 0..15
        const int d0 = (t & 15) * 8;            // 8 dims per thread
        const size_t hrow = (size_t)head_idx[tb + p] * ES;
        const size_t trow = (size_t)tail_idx[tb + p] * ES;
#pragma unroll
        for (int u = 0; u < 8; ++u) {
            float hv = ent_emb[hrow + d0 + u];
            float tv = ent_emb[trow + d0 + u];
            sHE[p * ES + d0 + u] = f2bf(hv);
            sTE[p * ES + d0 + u] = f2bf(tv);
            sWR[p * ES + d0 + u] = f2bf(tv - hv);
        }
    }
    __syncthreads();

    // ---------------- Phase 1: ws[16x128] = weak_rel @ W_bil^T (WMMA bf16) ----------------
    {
        const int n0 = wave * 16;
        v8f c = {};
#pragma unroll
        for (int k0 = 0; k0 < ES; k0 += 32) {
            v16bf a  = load_a(sWR, k0, lane);
            v16bf bm = load_b(sW, k0, n0, lane);
            c = __builtin_amdgcn_wmma_f32_16x16x32_bf16(false, a, false, bm,
                                                        (short)0, c, false, false);
        }
        const int n  = n0 + (lane & 15);
        const int mh = (lane & 16) ? 8 : 0;
#pragma unroll
        for (int r = 0; r < 8; ++r) sWS[(r + mh) * ES + n] = c[r];
    }
    __syncthreads();

    // ---------------- Phase 2: neighbor attention (32 tasks, 4 per wave) ----------------
    for (int q = 0; q < 4; ++q) {
        const int tt  = wave * 4 + q;
        const int p   = tt >> 1;                // pair in tile
        const int s   = tt & 1;                 // 0=head side, 1=tail side
        const int idx = sIDX[s * TILE + p];

        const int k1 = lane;                    // always < 50
        const int k2 = lane + 32;               // valid if lane < 18
        const int nb1 = conn_rel[(size_t)idx * KNB + k1];
        sNB[wave * 64 + k1] = nb1;
        int nb2 = 0;
        if (k2 < KNB) { nb2 = conn_rel[(size_t)idx * KNB + k2]; sNB[wave * 64 + k2] = nb2; }

        const float* wsr = sWS + p * ES;
        float sc1, sc2 = -3.402823466e38f;
        {
            const float* rr = rel_emb + (size_t)nb1 * ES;
            float acc = 0.f;
#pragma unroll 4
            for (int j = 0; j < ES; j += 4) {
                float4 a = *(const float4*)(rr + j);
                float4 w = *(const float4*)(wsr + j);
                acc += a.x * w.x + a.y * w.y + a.z * w.z + a.w * w.w;
            }
            sc1 = acc;
        }
        if (k2 < KNB) {
            const float* rr = rel_emb + (size_t)nb2 * ES;
            float acc = 0.f;
#pragma unroll 4
            for (int j = 0; j < ES; j += 4) {
                float4 a = *(const float4*)(rr + j);
                float4 w = *(const float4*)(wsr + j);
                acc += a.x * w.x + a.y * w.y + a.z * w.z + a.w * w.w;
            }
            sc2 = acc;
        }
        // wave-wide softmax over the (up to) 50 valid scores
        float m = fmaxf(sc1, sc2);
        for (int off = 16; off; off >>= 1) m = fmaxf(m, __shfl_xor(m, off, 32));
        float e1 = __expf(sc1 - m);
        float e2 = (k2 < KNB) ? __expf(sc2 - m) : 0.f;
        float sum = e1 + e2;
        for (int off = 16; off; off >>= 1) sum += __shfl_xor(sum, off, 32);
        const float inv = 1.f / sum;
        sATT[wave * 64 + k1] = e1 * inv;
        if (k2 < KNB) sATT[wave * 64 + k2] = e2 * inv;

        // column-parallel weighted aggregate: lane owns 4 output dims
        const int col = lane * 4;
        float ax = 0.f, ay = 0.f, az = 0.f, aw = 0.f;
#pragma unroll 2
        for (int k = 0; k < KNB; ++k) {
            float a = sATT[wave * 64 + k];
            const float4 v = *(const float4*)(rel_emb + (size_t)sNB[wave * 64 + k] * ES + col);
            ax += a * v.x; ay += a * v.y; az += a * v.z; aw += a * v.w;
        }
        bf16* dst = (s ? sTA : sHA) + p * ES + col;
        dst[0] = f2bf(ax); dst[1] = f2bf(ay); dst[2] = f2bf(az); dst[3] = f2bf(aw);
    }
    __syncthreads();

    // ---------------- Phase 3: relu(agg@W1^T + ent@W2^T + b1 + b2) (WMMA bf16) ----------------
    {
        const int n0 = wave * 16;
        const int n  = n0 + (lane & 15);
        const int mh = (lane & 16) ? 8 : 0;
        const float bsum = b1[n] + b2[n];
        const bf16* sW1 = sW + ES * ES;
        const bf16* sW2 = sW + 2 * ES * ES;

        // head side
        {
            v8f c = {};
#pragma unroll
            for (int k0 = 0; k0 < ES; k0 += 32) {
                v16bf a  = load_a(sHA, k0, lane);
                v16bf bm = load_b(sW1, k0, n0, lane);
                c = __builtin_amdgcn_wmma_f32_16x16x32_bf16(false, a, false, bm,
                                                            (short)0, c, false, false);
            }
#pragma unroll
            for (int k0 = 0; k0 < ES; k0 += 32) {
                v16bf a  = load_a(sHE, k0, lane);
                v16bf bm = load_b(sW2, k0, n0, lane);
                c = __builtin_amdgcn_wmma_f32_16x16x32_bf16(false, a, false, bm,
                                                            (short)0, c, false, false);
            }
#pragma unroll
            for (int r = 0; r < 8; ++r) {
                float v = c[r] + bsum;
                v = fmaxf(v, 0.f);
                out[(size_t)(tb + r + mh) * (2 * ES) + n] = v;   // side 0
            }
        }
        // tail side
        {
            v8f c = {};
#pragma unroll
            for (int k0 = 0; k0 < ES; k0 += 32) {
                v16bf a  = load_a(sTA, k0, lane);
                v16bf bm = load_b(sW1, k0, n0, lane);
                c = __builtin_amdgcn_wmma_f32_16x16x32_bf16(false, a, false, bm,
                                                            (short)0, c, false, false);
            }
#pragma unroll
            for (int k0 = 0; k0 < ES; k0 += 32) {
                v16bf a  = load_a(sTE, k0, lane);
                v16bf bm = load_b(sW2, k0, n0, lane);
                c = __builtin_amdgcn_wmma_f32_16x16x32_bf16(false, a, false, bm,
                                                            (short)0, c, false, false);
            }
#pragma unroll
            for (int r = 0; r < 8; ++r) {
                float v = c[r] + bsum;
                v = fmaxf(v, 0.f);
                out[(size_t)(tb + r + mh) * (2 * ES) + ES + n] = v;  // side 1
            }
        }
    }
}

extern "C" void kernel_launch(void* const* d_in, const int* in_sizes, int n_in,
                              void* d_out, int out_size, void* d_ws, size_t ws_size,
                              hipStream_t stream)
{
    const float* ent_emb  = (const float*)d_in[0];
    const float* rel_emb  = (const float*)d_in[1];
    const float* W_bil    = (const float*)d_in[2];
    const float* W1       = (const float*)d_in[3];
    const float* b1       = (const float*)d_in[4];
    const float* W2       = (const float*)d_in[5];
    const float* b2       = (const float*)d_in[6];
    const int*   head_idx = (const int*)d_in[7];
    const int*   tail_idx = (const int*)d_in[8];
    const int*   conn_rel = (const int*)d_in[9];
    float*       out      = (float*)d_out;

    // LDS: 3*128*128 + 5*16*128 bf16  +  (16*128 + 8*64) f32  +  (8*64 + 32) int
    const size_t smem = (size_t)(3 * ES * ES + 5 * TILE * ES) * sizeof(bf16)
                      + (size_t)(TILE * ES + 8 * 64) * sizeof(float)
                      + (size_t)(8 * 64 + 32) * sizeof(int);

    koa_kernel<<<NPAIRS / TILE, THREADS, smem, stream>>>(
        ent_emb, rel_emb, W_bil, W1, b1, W2, b2, head_idx, tail_idx, conn_rel, out);
}
